// LSTM_Model_Phoneme_69999376990724
// MI455X (gfx1250) — matrix operand order repaired and checked by reading, not compiled, not run on hardware
//
#include <hip/hip_runtime.h>

// ---------------------------------------------------------------------------
// LSTM encoder-decoder (B=64, S=T=64, H=512, V=2048, L=2, F=4) for gfx1250.
// All GEMMs use v_wmma_f32_16x16x32_f16 (f16 operands, f32 accumulate).
// ---------------------------------------------------------------------------

#define BB 64
#define SS 64
#define TT 64
#define HH 512
#define VV 2048
#define FF 4
#define G4H 2048   // 4*H
#define DD 2048    // F*H (layer-0 input width)

typedef _Float16 v8h  __attribute__((ext_vector_type(8)));
typedef _Float16 v16h __attribute__((ext_vector_type(16)));
typedef float    v8f  __attribute__((ext_vector_type(8)));

union V16u { v16h v; v8h h[2]; };

__device__ __forceinline__ v8f wmma_f16(v16h a, v16h b, v8f c) {
  return __builtin_amdgcn_wmma_f32_16x16x32_f16(false, a, false, b, (short)0, c,
                                                false, false);
}

// A fragment: 16x32 f16, rows striped in lanes (M=lane&15), per ISA layout:
// lane<16: elems 0..7 -> K=k0..k0+7, 8..15 -> K=k0+16..k0+23; lane>=16: +8.
__device__ __forceinline__ v16h load_a(const _Float16* __restrict__ arow,
                                       int k0, int lane) {
  const int kb = (lane >> 4) << 3;
  V16u u;
  u.h[0] = *(const v8h*)(arow + k0 + kb);
  u.h[1] = *(const v8h*)(arow + k0 + 16 + kb);
  return u.v;
}

// B fragment: 32x16 f16 where B[k][n] = W[n0+n][k] (W is [N,K] row-major).
// lane<16: elems 0..7 -> K=k0..k0+7, 8..15 -> K=k0+16..k0+23; lane>=16: +8. N = lane&15.
__device__ __forceinline__ v16h load_b(const _Float16* __restrict__ W, int ldw,
                                       int n0, int k0, int lane) {
  const _Float16* p =
      W + (long)(n0 + (lane & 15)) * ldw + k0 + ((lane >> 4) << 3);
  V16u u;
  u.h[0] = *(const v8h*)p;
  u.h[1] = *(const v8h*)(p + 16);
  return u.v;
}

// ---------------- generic GEMM: C[M,N] = A[M,K] * W[N,K]^T + bias ----------
// block = 128 threads (4 waves), block tile 64x64, wave tile 16x64.
// grid = (N/64, M/64)
__global__ void __launch_bounds__(128)
gemm_f16f32(const _Float16* __restrict__ A, int lda,
            const _Float16* __restrict__ W, int ldw,
            const float* __restrict__ bias,
            float* __restrict__ C, int ldc, int K) {
  const int lane = threadIdx.x & 31;
  const int wave = threadIdx.x >> 5;
  const int row0 = blockIdx.y * 64 + wave * 16;
  const int n0 = blockIdx.x * 64;

  v8f acc[4] = {};
  const _Float16* arow = A + (long)(row0 + (lane & 15)) * lda;

  for (int k0 = 0; k0 < K; k0 += 32) {
    v16h a = load_a(arow, k0, lane);
#pragma unroll
    for (int nt = 0; nt < 4; ++nt) {
      v16h b = load_b(W, ldw, n0 + nt * 16, k0, lane);
      acc[nt] = wmma_f16(a, b, acc[nt]);
    }
  }

  const int mofs = (lane >> 4) << 3;  // upper half-lanes hold M=+8
#pragma unroll
  for (int nt = 0; nt < 4; ++nt) {
    const int col = n0 + nt * 16 + (lane & 15);
    const float bv = bias ? bias[col] : 0.0f;
#pragma unroll
    for (int r = 0; r < 8; ++r) {
      const int rowm = row0 + r + mofs;
      C[(long)rowm * ldc + col] = acc[nt][r] + bv;
    }
  }
}

// ---------------- LSTM recurrent step --------------------------------------
// grid.x = H/64 = 8 hidden chunks; block = 256 threads (8 waves).
// Each block computes gate tiles [64 batch x 64 hid] for ALL FOUR gates of
// its chunk (cols g*H + chunk*64 .. +63), so the cell update is block-local.
// Wave w: row strip rb=w&3 (16 rows), gates {w>>2, (w>>2)+2}.
__device__ __forceinline__ float sigm(float x) {
  return 1.0f / (1.0f + __expf(-x));
}

__global__ void __launch_bounds__(256)
lstm_step(const float* __restrict__ pre,       // [B*T, 4H], row = b*T + t
          const _Float16* __restrict__ hprev,  // row b at hprev + b*hps; null=0
          int hps,
          const _Float16* __restrict__ Whh,    // [4H, H] f16
          float* __restrict__ c,               // [B, H] in-place
          _Float16* __restrict__ hseq,         // [B*T, H]; write row b*T + t
          int t) {
  __shared__ float gbuf[4][64][64];
  const int lane = threadIdx.x & 31;
  const int wave = threadIdx.x >> 5;
  const int chunk = blockIdx.x;
  const int rb = wave & 3;
  const int g0 = wave >> 2;  // gates g0 and g0+2

  v8f acc[2][4] = {};
  if (hprev) {
    const _Float16* arow = hprev + (long)(rb * 16 + (lane & 15)) * hps;
    for (int k0 = 0; k0 < HH; k0 += 32) {
      v16h a = load_a(arow, k0, lane);
#pragma unroll
      for (int gi = 0; gi < 2; ++gi) {
        const int g = g0 + 2 * gi;
#pragma unroll
        for (int nt = 0; nt < 4; ++nt) {
          v16h b = load_b(Whh, HH, g * HH + chunk * 64 + nt * 16, k0, lane);
          acc[gi][nt] = wmma_f16(a, b, acc[gi][nt]);
        }
      }
    }
  }

  const int mofs = (lane >> 4) << 3;
#pragma unroll
  for (int gi = 0; gi < 2; ++gi) {
    const int g = g0 + 2 * gi;
#pragma unroll
    for (int nt = 0; nt < 4; ++nt) {
      const int j = nt * 16 + (lane & 15);
#pragma unroll
      for (int r = 0; r < 8; ++r) {
        const int b = rb * 16 + r + mofs;
        gbuf[g][b][j] =
            acc[gi][nt][r] +
            pre[((long)b * TT + t) * G4H + g * HH + chunk * 64 + j];
      }
    }
  }
  __syncthreads();

  for (int idx = threadIdx.x; idx < 64 * 64; idx += 256) {
    const int b = idx >> 6;
    const int j = idx & 63;
    const int hc = chunk * 64 + j;
    const float iv = gbuf[0][b][j];
    const float fv = gbuf[1][b][j];
    const float gv = gbuf[2][b][j];
    const float ov = gbuf[3][b][j];
    const float cc = sigm(fv) * c[b * HH + hc] + sigm(iv) * tanhf(gv);
    c[b * HH + hc] = cc;
    const float hh = sigm(ov) * tanhf(cc);
    hseq[((long)b * TT + t) * HH + hc] = (_Float16)hh;
  }
}

// ---------------- small helper kernels -------------------------------------
__global__ void zero_f32(float* __restrict__ p, long n) {
  for (long i = blockIdx.x * (long)blockDim.x + threadIdx.x; i < n;
       i += (long)gridDim.x * blockDim.x)
    p[i] = 0.0f;
}

__global__ void cvt_f16(const float* __restrict__ s, _Float16* __restrict__ d,
                        long n) {
  for (long i = blockIdx.x * (long)blockDim.x + threadIdx.x; i < n;
       i += (long)gridDim.x * blockDim.x)
    d[i] = (_Float16)s[i];
}

__global__ void bias_sum(const float* __restrict__ a,
                         const float* __restrict__ b, float* __restrict__ o,
                         int n) {
  int i = blockIdx.x * blockDim.x + threadIdx.x;
  if (i < n) o[i] = a[i] + b[i];
}

// embeddings: out[(b*S+s), f*H + h] = (f16) emb[f][tok][h]
__global__ void embed_enc(const int* __restrict__ x,
                          const float* __restrict__ emb,
                          _Float16* __restrict__ out) {
  const long total = (long)BB * SS * FF * HH;
  for (long i = blockIdx.x * (long)blockDim.x + threadIdx.x; i < total;
       i += (long)gridDim.x * blockDim.x) {
    const int h = (int)(i % HH);
    const long r = i / HH;
    const int f = (int)(r % FF);
    const long bs = r / FF;
    const int tok = x[bs * FF + f];
    out[bs * DD + f * HH + h] = (_Float16)emb[((long)f * VV + tok) * HH + h];
  }
}

// decoder tokens: t==0 -> BOS(f==0)/PAD else labels[b, t-1, f]
__global__ void embed_dec(const int* __restrict__ labels,
                          const float* __restrict__ emb,
                          _Float16* __restrict__ out) {
  const long total = (long)BB * TT * FF * HH;
  for (long i = blockIdx.x * (long)blockDim.x + threadIdx.x; i < total;
       i += (long)gridDim.x * blockDim.x) {
    const int h = (int)(i % HH);
    const long r = i / HH;
    const int f = (int)(r % FF);
    const long bt = r / FF;
    const int b = (int)(bt / TT);
    const int t = (int)(bt % TT);
    const int tok =
        (t == 0) ? ((f == 0) ? 1 : 0) : labels[((long)b * TT + (t - 1)) * FF + f];
    out[bt * DD + f * HH + h] = (_Float16)emb[((long)f * VV + tok) * HH + h];
  }
}

// per-(b,t,f) row log-softmax + NLL; loss += nll / (B*T)
__global__ void __launch_bounds__(256)
ce_loss(const float* __restrict__ outs, const int* __restrict__ labels,
        float* __restrict__ loss) {
  __shared__ float red[256];
  const long r = blockIdx.x;
  const float* row = outs + r * VV;

  float mx = -3.4e38f;
  for (int v = threadIdx.x; v < VV; v += 256) mx = fmaxf(mx, row[v]);
  red[threadIdx.x] = mx;
  __syncthreads();
  for (int s = 128; s > 0; s >>= 1) {
    if (threadIdx.x < s)
      red[threadIdx.x] = fmaxf(red[threadIdx.x], red[threadIdx.x + s]);
    __syncthreads();
  }
  mx = red[0];
  __syncthreads();

  float sum = 0.0f;
  for (int v = threadIdx.x; v < VV; v += 256) sum += __expf(row[v] - mx);
  red[threadIdx.x] = sum;
  __syncthreads();
  for (int s = 128; s > 0; s >>= 1) {
    if (threadIdx.x < s) red[threadIdx.x] += red[threadIdx.x + s];
    __syncthreads();
  }
  if (threadIdx.x == 0) {
    const float logZ = mx + __logf(red[0]);
    const int lab = labels[r];
    atomicAdd(loss, (logZ - row[lab]) * (1.0f / (float)(BB * TT)));
  }
}

// ---------------------------------------------------------------------------
extern "C" void kernel_launch(void* const* d_in, const int* in_sizes, int n_in,
                              void* d_out, int out_size, void* d_ws,
                              size_t ws_size, hipStream_t stream) {
  if (n_in < 22) return;

  const int* x = (const int*)d_in[0];
  const int* labels = (const int*)d_in[1];
  const float* emb_enc = (const float*)d_in[2];
  const float* emb_dec = (const float*)d_in[11];
  const float* heads_W = (const float*)d_in[20];
  const float* heads_b = (const float*)d_in[21];

  // Disambiguate pytree ordering of {Wih,Whh,bih,bhh}: layer-0 Wih is
  // 4H*(F*H)=4194304 elems, Whh is 4H*H=1048576. bih/bhh order irrelevant.
  const bool wihFirst = (in_sizes[3] == 4 * HH * FF * HH);
  const int iW = wihFirst ? 0 : 1;
  const int iU = wihFirst ? 1 : 0;
  const int bases[4] = {3, 7, 12, 16};  // enc0, enc1, dec0, dec1
  const float *Wih[4], *Whh[4], *b0[4], *b1[4];
  for (int i = 0; i < 4; ++i) {
    Wih[i] = (const float*)d_in[bases[i] + iW];
    Whh[i] = (const float*)d_in[bases[i] + iU];
    b0[i] = (const float*)d_in[bases[i] + 2];
    b1[i] = (const float*)d_in[bases[i] + 3];
  }
  const long wihN[4] = {(long)G4H * DD, (long)G4H * HH, (long)G4H * DD,
                        (long)G4H * HH};
  const int wihLd[4] = {DD, HH, DD, HH};

  // ---- workspace bump allocator ----
  char* wp = (char*)d_ws;
  auto alloc = [&](size_t bytes) -> void* {
    void* p = (void*)wp;
    wp += (bytes + 255) & ~(size_t)255;
    return p;
  };
  _Float16* enc_in = (_Float16*)alloc((size_t)BB * SS * DD * 2);
  _Float16* dec_in = (_Float16*)alloc((size_t)BB * TT * DD * 2);
  _Float16* hseq[4];  // enc0, enc1, dec0, dec1
  for (int i = 0; i < 4; ++i) hseq[i] = (_Float16*)alloc((size_t)BB * TT * HH * 2);
  float* pre = (float*)alloc((size_t)BB * TT * G4H * 4);
  float* cst[2] = {(float*)alloc((size_t)BB * HH * 4),
                   (float*)alloc((size_t)BB * HH * 4)};
  float* bsum[4];
  for (int i = 0; i < 4; ++i) bsum[i] = (float*)alloc(G4H * 4);
  _Float16 *wih16[4], *whh16[4];
  for (int i = 0; i < 4; ++i) {
    wih16[i] = (_Float16*)alloc((size_t)wihN[i] * 2);
    whh16[i] = (_Float16*)alloc((size_t)G4H * HH * 2);
  }
  _Float16* headsW16 = (_Float16*)alloc((size_t)FF * VV * HH * 2);
  (void)ws_size;

  float* outs = (float*)d_out;                       // [B,T,F,V]
  float* lossp = outs + (long)BB * TT * FF * VV;     // scalar
  (void)out_size;

  // ---- init ----
  zero_f32<<<1, 64, 0, stream>>>(lossp, 1);
  zero_f32<<<128, 256, 0, stream>>>(cst[0], (long)BB * HH);
  zero_f32<<<128, 256, 0, stream>>>(cst[1], (long)BB * HH);

  // ---- weight conversions (fp32 -> f16) ----
  for (int i = 0; i < 4; ++i) {
    cvt_f16<<<2048, 256, 0, stream>>>(Wih[i], wih16[i], wihN[i]);
    cvt_f16<<<1024, 256, 0, stream>>>(Whh[i], whh16[i], (long)G4H * HH);
    bias_sum<<<(G4H + 255) / 256, 256, 0, stream>>>(b0[i], b1[i], bsum[i], G4H);
  }
  cvt_f16<<<2048, 256, 0, stream>>>(heads_W, headsW16, (long)FF * VV * HH);

  // ---- embeddings ----
  embed_enc<<<4096, 256, 0, stream>>>(x, emb_enc, enc_in);
  embed_dec<<<4096, 256, 0, stream>>>(labels, emb_dec, dec_in);

  const int MM = BB * TT;  // 4096 rows for all sequence GEMMs
  auto run_layer = [&](const _Float16* xin, int ldx, int Kin, int inst,
                       const _Float16* h0, float* cbuf, _Float16* hs) {
    // pre = xin @ Wih^T + (bih+bhh)   -> [4096, 4H]
    gemm_f16f32<<<dim3(G4H / 64, MM / 64), 128, 0, stream>>>(
        xin, ldx, wih16[inst], Kin, bsum[inst], pre, G4H, Kin);
    // recurrence
    for (int t = 0; t < TT; ++t) {
      const _Float16* hp =
          (t == 0) ? h0 : (const _Float16*)(hs + (long)(t - 1) * HH);
      lstm_step<<<HH / 64, 256, 0, stream>>>(pre, hp, TT * HH, whh16[inst],
                                             cbuf, hs, t);
    }
  };

  // ---- encoder (h0 = c0 = 0) ----
  run_layer(enc_in, DD, DD, 0, nullptr, cst[0], hseq[0]);
  run_layer(hseq[0], HH, HH, 1, nullptr, cst[1], hseq[1]);

  // ---- decoder (h0/c0 = encoder finals; c continues in-place) ----
  const _Float16* hfin0 = hseq[0] + (long)(SS - 1) * HH;  // row b: + b*T*H
  const _Float16* hfin1 = hseq[1] + (long)(SS - 1) * HH;
  run_layer(dec_in, DD, DD, 2, hfin0, cst[0], hseq[2]);
  run_layer(hseq[2], HH, HH, 3, hfin1, cst[1], hseq[3]);

  // ---- heads: outs[4096, F*V] = dec_out @ heads_W^T + heads_b ----
  gemm_f16f32<<<dim3((FF * VV) / 64, MM / 64), 128, 0, stream>>>(
      hseq[3], HH, headsW16, HH, heads_b, outs, FF * VV, HH);

  // ---- loss ----
  ce_loss<<<MM * FF, 256, 0, stream>>>(outs, labels, lossp);
}
